// GraphAttentionLayer_80650895884363
// MI455X (gfx1250) — matrix operand order repaired and checked by reading, not compiled, not run on hardware
//
#include <hip/hip_runtime.h>
#include <hip/hip_bf16.h>
#include <hip/hip_fp16.h>

// ---------------------------------------------------------------------------
// GAT layer, MI455X (gfx1250, wave32, WMMA)
//   N=10000 nodes, F_IN=512, F_OUT=256
//   Wh      = h @ W                       (fp32 WMMA 16x16x4, exact)
//   e       = leaky_relu(s1[i] + s2[j])
//   logits  = adj ? e : -9e15 ; row softmax fused into A@V (two-pass flash)
//   h_prime = softmax(logits) @ Wh        (fp16 WMMA 16x16x32, fp32 accum)
//   out     = elu(h_prime)
// ---------------------------------------------------------------------------

#define N_NODES 10000
#define NPAD    10048          // 157 * 64 : K padding for 64-wide attention tiles
#define F_IN    512
#define F_OUT   256
#define ALPHA   0.2f
#define NEG_INF -9.0e15f

typedef __attribute__((ext_vector_type(2)))  float    v2f;
typedef __attribute__((ext_vector_type(8)))  float    v8f;
typedef __attribute__((ext_vector_type(4)))  _Float16 v4h;
typedef __attribute__((ext_vector_type(8)))  _Float16 v8h;
typedef __attribute__((ext_vector_type(16))) _Float16 v16h;

// ---------------------------------------------------------------------------
// Kernel 1: Wh = h @ W, exact fp32 via V_WMMA_F32_16X16X4_F32.
// One wave per 16x32 output tile (two accumulators -> independent WMMA
// streams, A fragment reused). N_NODES % 16 == 0, so padding tiles are
// uniformly out-of-range: block-level early-out, branch-free hot loop.
// ---------------------------------------------------------------------------
__global__ __launch_bounds__(32)
void wh_gemm_f32wmma(const float* __restrict__ h, const float* __restrict__ W,
                     float* __restrict__ Wh) {
  const int row0 = blockIdx.x * 16;
  const int col0 = blockIdx.y * 32;
  const int lane = threadIdx.x & 31;
  const int half = lane >> 4;
  const int m16  = lane & 15;

  v8f acc0 = {};
  v8f acc1 = {};

  if (row0 < N_NODES) {                       // whole tile valid (uniform test)
#if __has_builtin(__builtin_amdgcn_wmma_f32_16x16x4_f32)
    const float* hrow = h + (size_t)(row0 + m16) * F_IN + half * 2;
    const float* w0   = W + col0 + m16;
    const float* w1   = W + col0 + 16 + m16;
    for (int k = 0; k < F_IN; k += 4) {
      const v2f a = *(const v2f*)(hrow + k);                 // global_load_b64
      const size_t kb = (size_t)(k + half * 2) * F_OUT;
      v2f b0, b1;
      b0.x = w0[kb]; b0.y = w0[kb + F_OUT];
      b1.x = w1[kb]; b1.y = w1[kb + F_OUT];
      acc0 = __builtin_amdgcn_wmma_f32_16x16x4_f32(false, a, false, b0,
                                                   (short)0, acc0, false, false);
      acc1 = __builtin_amdgcn_wmma_f32_16x16x4_f32(false, a, false, b1,
                                                   (short)0, acc1, false, false);
    }
#else
    for (int r = 0; r < 8; ++r) {
      const int arow = row0 + r + half * 8;
      const float* hrow = h + (size_t)arow * F_IN;
      float s0 = 0.0f, s1v = 0.0f;
      for (int k = 0; k < F_IN; ++k) {
        s0  += hrow[k] * W[(size_t)k * F_OUT + col0 + m16];
        s1v += hrow[k] * W[(size_t)k * F_OUT + col0 + 16 + m16];
      }
      acc0[r] = s0; acc1[r] = s1v;
    }
#endif
  }

  float* o0 = Wh + (size_t)row0 * F_OUT + col0 + m16;
#pragma unroll
  for (int r = 0; r < 8; ++r) {
    o0[(size_t)(r + half * 8) * F_OUT]      = acc0[r];
    o0[(size_t)(r + half * 8) * F_OUT + 16] = acc1[r];
  }
}

// ---------------------------------------------------------------------------
// Kernel 2: s1 = Wh @ a[:256], s2 = Wh @ a[256:]. One wave per row.
// ---------------------------------------------------------------------------
__global__ __launch_bounds__(256)
void score_proj(const float* __restrict__ Wh, const float* __restrict__ a,
                float* __restrict__ s1, float* __restrict__ s2) {
  const int row  = blockIdx.x * 8 + threadIdx.y;   // < NPAD
  const int lane = threadIdx.x;                    // 0..31
  const float* wr = Wh + (size_t)row * F_OUT;
  float d1 = 0.0f, d2 = 0.0f;
#pragma unroll
  for (int c = lane; c < F_OUT; c += 32) {
    const float v = wr[c];
    d1 += v * a[c];
    d2 += v * a[F_OUT + c];
  }
#pragma unroll
  for (int off = 16; off > 0; off >>= 1) {
    d1 += __shfl_xor(d1, off, 32);
    d2 += __shfl_xor(d2, off, 32);
  }
  if (lane == 0) { s1[row] = d1; s2[row] = d2; }
}

// ---------------------------------------------------------------------------
// Kernel 3: pack Wh into column-major fp16 (K contiguous) for WMMA B-operand.
// ---------------------------------------------------------------------------
__global__ __launch_bounds__(256)
void pack_whT_f16(const float* __restrict__ Wh, _Float16* __restrict__ WhT) {
  const int r = blockIdx.x * 32 + threadIdx.x;   // node index, < NPAD
  const int c = blockIdx.y * 8 + threadIdx.y;    // feature, < F_OUT
  WhT[(size_t)c * NPAD + r] = (_Float16)Wh[(size_t)r * F_OUT + c];
}

// ---------------------------------------------------------------------------
// Kernel 4: per-row softmax stats (max, sum), online accumulation.
// One 256-thread block streams the 40 KB adj row with global_prefetch.
// exp(logit - m) reproduces the reference's -9e15 masking exactly,
// including the all-masked-row -> uniform case.
// ---------------------------------------------------------------------------
__global__ __launch_bounds__(256)
void row_softmax_stats(const int* __restrict__ adj, const float* __restrict__ s1,
                       const float* __restrict__ s2, float* __restrict__ rowmax,
                       float* __restrict__ rowsum) {
  __shared__ float sm[256];
  __shared__ float sl[256];
  const int row = blockIdx.x;
  const int t   = threadIdx.x;
  const int* arow = adj + (size_t)row * N_NODES;
  const float s1r = s1[row];

  float m = -__builtin_inff();
  float l = 0.0f;
  for (int j = t; j < N_NODES; j += 256) {
    __builtin_prefetch(arow + j + 1024, 0, 0);   // global_prefetch_b8
    float e = s1r + s2[j];
    e = (e >= 0.0f) ? e : ALPHA * e;
    const float logit = (arow[j] > 0) ? e : NEG_INF;
    if (logit > m) {
      l = l * __expf(m - logit) + 1.0f;
      m = logit;
    } else {
      l += __expf(logit - m);
    }
  }
  sm[t] = m; sl[t] = l;
  __syncthreads();
  for (int s = 128; s > 0; s >>= 1) {
    if (t < s) {
      const float ma = sm[t], mb = sm[t + s];
      const float mx = fmaxf(ma, mb);
      sl[t] = sl[t] * __expf(ma - mx) + sl[t + s] * __expf(mb - mx);
      sm[t] = mx;
    }
    __syncthreads();
  }
  if (t == 0) { rowmax[row] = sm[0]; rowsum[row] = sl[0]; }
}

// ---------------------------------------------------------------------------
// Kernel 5: h_prime = softmax(logits) @ Wh, fused, + ELU.
// Block = 256 threads (8 waves) owns 16 output rows x all 256 cols.
// Per 64-wide K tile: each thread reads its 4 adj entries with ONE int4
// (global_load_b128), computes 4 exp weights, writes them with ONE
// ds_store_b64. Each wave then runs 2 K-substeps x 2 col-tiles = 4
// v_wmma_f32_16x16x32_f16 per barrier, A via ds_read_b128, B via
// global_load_b128 from column-major Wh_f16. adj is read exactly once here.
// ---------------------------------------------------------------------------
__global__ __launch_bounds__(256)
void attn_av_gemm(const int* __restrict__ adj, const float* __restrict__ s1,
                  const float* __restrict__ s2, const float* __restrict__ rowmax,
                  const float* __restrict__ rowsum,
                  const _Float16* __restrict__ WhT, float* __restrict__ out) {
  __shared__ _Float16 Plds[2][16][64];           // double-buffered P tile (4 KB)

  const int row0 = blockIdx.x * 16;
  const int t    = threadIdx.x;
  const int lane = t & 31;
  const int wave = t >> 5;                       // 0..7 -> 32-col slab
  const int half = lane >> 4;
  const int n16  = lane & 15;

  // P-build assignment: thread t covers (row pm, K pk4..pk4+3)
  const int pm  = t & 15;
  const int pk4 = (t >> 4) * 4;
  const float s1r  = s1[row0 + pm];
  const float mrow = rowmax[row0 + pm];
  const int*  arow = adj + (size_t)(row0 + pm) * N_NODES;

  const int c0 = wave * 32;
  const _Float16* b0p = WhT + (size_t)(c0 +      n16) * NPAD + half * 16;
  const _Float16* b1p = WhT + (size_t)(c0 + 16 + n16) * NPAD + half * 16;

  v8f acc0 = {};
  v8f acc1 = {};

  for (int k0 = 0, it = 0; k0 < NPAD; k0 += 64, ++it) {
    const int buf = it & 1;

    // ---- build 16x64 exp-weight tile into LDS ----
    v4h pv;
    if (k0 + 64 <= N_NODES) {                    // fast path: one b128 adj read
      const int4 av = *(const int4*)(arow + k0 + pk4);
      __builtin_prefetch(arow + k0 + 128 + pk4, 0, 0);
      const int avi[4] = {av.x, av.y, av.z, av.w};
#pragma unroll
      for (int q = 0; q < 4; ++q) {
        float e = s1r + s2[k0 + pk4 + q];
        e = (e >= 0.0f) ? e : ALPHA * e;
        pv[q] = (_Float16)__expf(((avi[q] > 0) ? e : NEG_INF) - mrow);
      }
    } else {                                     // tail tile with K padding
#pragma unroll
      for (int q = 0; q < 4; ++q) {
        const int j = k0 + pk4 + q;
        float p = 0.0f;
        if (j < N_NODES) {
          float e = s1r + s2[j];
          e = (e >= 0.0f) ? e : ALPHA * e;
          p = __expf(((arow[j] > 0) ? e : NEG_INF) - mrow);
        }
        pv[q] = (_Float16)p;
      }
    }
    *(v4h*)&Plds[buf][pm][pk4] = pv;             // single ds_store_b64
    __syncthreads();

#pragma unroll
    for (int sub = 0; sub < 2; ++sub) {
      // A operand: 16-bit 16x32 layout, two ds_read_b128 per lane
      const v8h alo = *(const v8h*)&Plds[buf][n16][sub * 32 + half * 16];
      const v8h ahi = *(const v8h*)&Plds[buf][n16][sub * 32 + half * 16 + 8];
      v16h a;
#pragma unroll
      for (int i = 0; i < 8; ++i) { a[i] = alo[i]; a[8 + i] = ahi[i]; }

      // B operands: column-major Wh_f16, K contiguous (2x b128 each)
      const v16h b0 = *(const v16h*)(b0p + k0 + sub * 32);
      const v16h b1 = *(const v16h*)(b1p + k0 + sub * 32);

      acc0 = __builtin_amdgcn_wmma_f32_16x16x32_f16(false, a, false, b0,
                                                    (short)0, acc0, false, false);
      acc1 = __builtin_amdgcn_wmma_f32_16x16x32_f16(false, a, false, b1,
                                                    (short)0, acc1, false, false);
    }
  }

  // ---- epilogue: attention normalization + ELU ----
#pragma unroll
  for (int r = 0; r < 8; ++r) {
    const int orow  = row0 + r + half * 8;       // < N_NODES (625*16 = 10000)
    const float inv = 1.0f / rowsum[orow];
    float v0 = acc0[r] * inv;
    float v1 = acc1[r] * inv;
    v0 = (v0 > 0.0f) ? v0 : expm1f(v0);
    v1 = (v1 > 0.0f) ? v1 : expm1f(v1);
    float* orp = out + (size_t)orow * F_OUT;
    orp[c0 + n16]      = v0;
    orp[c0 + 16 + n16] = v1;
  }
}

// ---------------------------------------------------------------------------
// Host-side launch. Inputs (setup_inputs order): h, adj, W, a.
// Workspace: Wh fp32 (10.3 MB) | WhT fp16 (5.1 MB) | s1,s2,rowmax,rowsum.
// ---------------------------------------------------------------------------
extern "C" void kernel_launch(void* const* d_in, const int* in_sizes, int n_in,
                              void* d_out, int out_size, void* d_ws, size_t ws_size,
                              hipStream_t stream) {
  (void)in_sizes; (void)n_in; (void)out_size; (void)ws_size;

  const float* h   = (const float*)d_in[0];
  const int*   adj = (const int*)d_in[1];
  const float* W   = (const float*)d_in[2];
  const float* a   = (const float*)d_in[3];
  float*       out = (float*)d_out;

  char* ws = (char*)d_ws;
  float*    Wh   = (float*)ws;                                        // NPAD*F_OUT f32
  _Float16* WhT  = (_Float16*)(ws + (size_t)NPAD * F_OUT * 4);        // F_OUT*NPAD f16
  float*    s1   = (float*)(ws + (size_t)NPAD * F_OUT * 4 + (size_t)F_OUT * NPAD * 2);
  float*    s2   = s1 + NPAD;
  float*    rmax = s2 + NPAD;
  float*    rsum = rmax + NPAD;

  wh_gemm_f32wmma<<<dim3(NPAD / 16, F_OUT / 32), 32, 0, stream>>>(h, W, Wh);
  score_proj<<<dim3(NPAD / 8), dim3(32, 8), 0, stream>>>(Wh, a, s1, s2);
  pack_whT_f16<<<dim3(NPAD / 32, F_OUT / 8), dim3(32, 8), 0, stream>>>(Wh, WhT);
  row_softmax_stats<<<N_NODES, 256, 0, stream>>>(adj, s1, s2, rmax, rsum);
  attn_av_gemm<<<N_NODES / 16, 256, 0, stream>>>(adj, s1, s2, rmax, rsum, WhT, out);
}
